// GVAE_52020643889430
// MI455X (gfx1250) — matrix operand order repaired and checked by reading, not compiled, not run on hardware
//
#include <hip/hip_runtime.h>
#include <hip/hip_bf16.h>
#include <stddef.h>

// ---------------- problem constants (match reference) ----------------
#define NN   102400   // nodes
#define EE   204800   // edges
#define BB   4096     // graphs
#define NPGC 25       // nodes per graph
#define HH   4        // heads
#define CC   64       // channels per head / embedding
#define HC   256      // H*C
#define EDIM 11       // edge feature dim
#define LATD 128      // latent
#define DHD  512      // decoder hidden
#define AOUT 500      // atom logits per graph
#define EOUT 6125     // triu logits per graph

typedef __attribute__((ext_vector_type(16))) __bf16    v16bf;
typedef __attribute__((ext_vector_type(8)))  float     v8f;
typedef __attribute__((ext_vector_type(4)))  unsigned  u32x4;

struct frag2 { u32x4 lo, hi; };

__device__ __forceinline__ __bf16 f2bf(float f) {
  unsigned u = __float_as_uint(f);
  unsigned r = u + 0x7FFFu + ((u >> 16) & 1u);   // round-to-nearest-even
  return __builtin_bit_cast(__bf16, (unsigned short)(r >> 16));
}
__device__ __forceinline__ __bf16 bfz() {
  return __builtin_bit_cast(__bf16, (unsigned short)0);
}
// A fragment: elements 0..7 at p[0..7], 8..15 at p[16..23]  (ISA 7.12.2 A layout)
__device__ __forceinline__ v16bf load_frag_a(const __bf16* p) {
  frag2 f;
  f.lo = *(const u32x4*)(p);
  f.hi = *(const u32x4*)(p + 16);
  return __builtin_bit_cast(v16bf, f);
}
// B fragment: elements 0..7 at p[0..7], 8..15 at p[16..23] (col-major W, same K order as A)
__device__ __forceinline__ v16bf load_frag_b(const __bf16* p) {
  frag2 f;
  f.lo = *(const u32x4*)(p);
  f.hi = *(const u32x4*)(p + 16);
  return __builtin_bit_cast(v16bf, f);
}
__device__ __forceinline__ unsigned fkey(float f) {   // order-preserving f32->u32
  unsigned u = __float_as_uint(f);
  return (u & 0x80000000u) ? ~u : (u | 0x80000000u);
}
__device__ __forceinline__ float kdec(unsigned k) {
  unsigned u = (k & 0x80000000u) ? (k ^ 0x80000000u) : ~k;
  return __uint_as_float(u);
}
__device__ __forceinline__ float sigm(float x) { return 1.0f / (1.0f + __expf(-x)); }

// ---------------- operand conversion kernels ----------------
// f32 [M,K] row-major -> bf16 [M,Kp] row-major, zero-padded K..Kp
__global__ void cvt_rows(const float* __restrict__ src, __bf16* __restrict__ dst,
                         long M, int K, int Kp) {
  long i = blockIdx.x * (long)blockDim.x + threadIdx.x;
  if (i >= M * Kp) return;
  long r = i / Kp; int k = (int)(i - r * Kp);
  dst[i] = (k < K) ? f2bf(src[r * K + k]) : bfz();
}
// f32 [K,N] row-major -> bf16 [N,Kp] (transpose to col-major, zero-padded)
__global__ void cvt_wT(const float* __restrict__ src, __bf16* __restrict__ dst,
                       int K, int N, int Kp) {
  long i = blockIdx.x * (long)blockDim.x + threadIdx.x;
  if (i >= (long)N * Kp) return;
  int n = (int)(i / Kp); int k = (int)(i - (long)n * Kp);
  dst[i] = (k < K) ? f2bf(src[(size_t)k * N + n]) : bfz();
}

// ---------------- WMMA tile GEMM ----------------
// C[M,N] = act(Abf[M,Kp] @ Wbf + bias (+C));  Wbf col-major [N][Kp], Kp%32==0, M%16==0
// flags: bit0 = relu, bit1 = accumulate into C
__global__ void __launch_bounds__(128)
gemm_bf16(const __bf16* __restrict__ A, const __bf16* __restrict__ W,
          const float* __restrict__ bias, float* __restrict__ Cm,
          int M, int N, int Kp, int flags)
{
  const int lane   = threadIdx.x & 31;
  const int wave   = threadIdx.x >> 5;
  const int tilesN = (N + 63) >> 6;                 // 16x64 strip per wave
  const int t      = blockIdx.x * 4 + wave;
  if (t >= (M >> 4) * tilesN) return;
  const int tM = t / tilesN, tN = t % tilesN;
  const int row0 = tM << 4, col0 = tN << 6;

  const int aK = (lane < 16) ? 0 : 8;               // A: lanes16-31 hold K=8..15,24..31
  const int bK = (lane < 16) ? 0 : 8;               // B: lanes16-31 hold K=8..15,24..31
  const int lc = lane & 15;

  const __bf16* Ap = A + (size_t)(row0 + lc) * Kp + aK;
  const __bf16* Bp[4];
#pragma unroll
  for (int nt = 0; nt < 4; ++nt) {
    int col = col0 + nt * 16 + lc;
    if (col > N - 1) col = N - 1;                   // clamp: garbage cols never stored
    Bp[nt] = W + (size_t)col * Kp + bK;
  }

  v8f acc[4] = {};
  for (int k0 = 0; k0 < Kp; k0 += 32) {
    v16bf af = load_frag_a(Ap + k0);
#pragma unroll
    for (int nt = 0; nt < 4; ++nt) {
      v16bf bf = load_frag_b(Bp[nt] + k0);
      acc[nt] = __builtin_amdgcn_wmma_f32_16x16x32_bf16(
          false, af, false, bf, (short)0, acc[nt], false, false);
    }
  }

  const int rB = (lane < 16) ? 0 : 8;               // C/D: M split at lane 16
#pragma unroll
  for (int nt = 0; nt < 4; ++nt) {
    int col = col0 + nt * 16 + lc;
    if (col < N) {
      float bv = bias ? bias[col] : 0.0f;
#pragma unroll
      for (int r = 0; r < 8; ++r) {
        float* p = Cm + (size_t)(row0 + rB + r) * N + col;
        float v = acc[nt][r] + bv;
        if (flags & 2) v += *p;
        if (flags & 1) v = fmaxf(v, 0.0f);
        *p = v;
      }
    }
  }
}

// ---------------- utility: zero u32 words ----------------
__global__ void zero_u32(unsigned* __restrict__ p, long n) {
  long i = blockIdx.x * (long)blockDim.x + threadIdx.x;
  long s = (long)gridDim.x * blockDim.x;
  for (; i < n; i += s) p[i] = 0u;
}

// ---------------- TransformerConv edge kernels ----------------
__global__ void edge_alpha(const float* __restrict__ q, const float* __restrict__ k,
                           const float* __restrict__ ea, const float* __restrict__ We,
                           const int* __restrict__ ei, float* __restrict__ alpha,
                           unsigned* __restrict__ amaxkey)
{
  int t = blockIdx.x * blockDim.x + threadIdx.x;
  if (t >= EE * HH) return;
  int e = t >> 2, h = t & 3;
  int src = ei[e], dst = ei[EE + e];
  float a[EDIM];
#pragma unroll
  for (int j = 0; j < EDIM; ++j) a[j] = ea[(size_t)e * EDIM + j];
  const float* qr = q + (size_t)dst * HC + h * CC;
  const float* kr = k + (size_t)src * HC + h * CC;
  const float* Wc = We + h * CC;                    // We[j, h*64+c], row-major [11,256]
  float acc = 0.0f;
  for (int c = 0; c < CC; ++c) {
    float ec = 0.0f;
#pragma unroll
    for (int j = 0; j < EDIM; ++j) ec += a[j] * Wc[j * HC + c];
    acc += qr[c] * (kr[c] + ec);
  }
  float al = acc * 0.125f;                          // 1/sqrt(64)
  alpha[t] = al;
  atomicMax(&amaxkey[dst * HH + h], fkey(al));
}

__global__ void edge_softmax(float* __restrict__ alpha, const unsigned* __restrict__ amaxkey,
                             float* __restrict__ denom, const int* __restrict__ ei)
{
  int t = blockIdx.x * blockDim.x + threadIdx.x;
  if (t >= EE * HH) return;
  int e = t >> 2, h = t & 3;
  int dst = ei[EE + e];
  unsigned kk = amaxkey[dst * HH + h];
  float m = (kk == 0u) ? 0.0f : kdec(kk);
  if (!(m == m) || fabsf(m) > 3.0e38f) m = 0.0f;    // isfinite guard
  float a = __expf(alpha[t] - m);
  alpha[t] = a;
  atomicAdd(&denom[dst * HH + h], a);
}

__global__ void edge_aggregate(const float* __restrict__ alpha, const float* __restrict__ denom,
                               const float* __restrict__ v, const float* __restrict__ ea,
                               const float* __restrict__ We, const int* __restrict__ ei,
                               float* __restrict__ outacc)
{
  int t = blockIdx.x * blockDim.x + threadIdx.x;
  if (t >= EE * HH) return;
  int e = t >> 2, h = t & 3;
  int src = ei[e], dst = ei[EE + e];
  float w = alpha[t] / (denom[dst * HH + h] + 1e-16f);
  float a[EDIM];
#pragma unroll
  for (int j = 0; j < EDIM; ++j) a[j] = ea[(size_t)e * EDIM + j];
  const float* vr = v + (size_t)src * HC + h * CC;
  const float* Wc = We + h * CC;
  float* od = outacc + (size_t)dst * HC + h * CC;
  for (int c = 0; c < CC; ++c) {
    float ec = 0.0f;
#pragma unroll
    for (int j = 0; j < EDIM; ++j) ec += a[j] * Wc[j * HC + c];
    atomicAdd(&od[c], w * (vr[c] + ec));
  }
}

// head-mean + gated skip (beta) + ReLU
__global__ void beta_combine(const float* __restrict__ outacc, const float* __restrict__ xr,
                             const float* __restrict__ Wb, float* __restrict__ hout)
{
  int n = blockIdx.x * blockDim.x + threadIdx.x;
  if (n >= NN) return;
  const float* oa = outacc + (size_t)n * HC;
  const float* xn = xr + (size_t)n * CC;
  float s = 0.0f;
  for (int c = 0; c < CC; ++c) {
    float om = 0.25f * (oa[c] + oa[CC + c] + oa[2 * CC + c] + oa[3 * CC + c]);
    float x  = xn[c];
    s += Wb[c] * om + Wb[CC + c] * x + Wb[2 * CC + c] * (om - x);
  }
  float beta = sigm(s);
  float* hn = hout + (size_t)n * CC;
  for (int c = 0; c < CC; ++c) {
    float om = 0.25f * (oa[c] + oa[CC + c] + oa[2 * CC + c] + oa[3 * CC + c]);
    float x  = xn[c];
    hn[c] = fmaxf(beta * x + (1.0f - beta) * om, 0.0f);   // relu(conv(...))
  }
}

// ---------------- BatchNorm (training stats, biased var) ----------------
__global__ void bn_stats(const float* __restrict__ h, float* __restrict__ stats) {
  __shared__ float ss[256], sq[256];
  int c = blockIdx.x;
  float s = 0.0f, s2 = 0.0f;
  for (int i = threadIdx.x; i < NN; i += 256) {
    float v = h[(size_t)i * CC + c];
    s += v; s2 += v * v;
  }
  ss[threadIdx.x] = s; sq[threadIdx.x] = s2;
  __syncthreads();
  for (int o = 128; o > 0; o >>= 1) {
    if (threadIdx.x < o) { ss[threadIdx.x] += ss[threadIdx.x + o]; sq[threadIdx.x] += sq[threadIdx.x + o]; }
    __syncthreads();
  }
  if (threadIdx.x == 0) {
    float mean = ss[0] / NN;
    float var  = sq[0] / NN - mean * mean;
    stats[c] = mean;
    stats[CC + c] = rsqrtf(var + 1e-5f);
  }
}
__global__ void bn_apply(float* __restrict__ h, const float* __restrict__ stats,
                         const float* __restrict__ g, const float* __restrict__ b) {
  long i = blockIdx.x * (long)blockDim.x + threadIdx.x;
  if (i >= (long)NN * CC) return;
  int c = (int)(i % CC);
  h[i] = (h[i] - stats[c]) * stats[CC + c] * g[c] + b[c];
}

// ---------------- Set2Set ----------------
__global__ void lstm_cell(const float* __restrict__ gates, float* __restrict__ hst,
                          float* __restrict__ cst) {
  int t = blockIdx.x * blockDim.x + threadIdx.x;
  if (t >= BB * CC) return;
  int b = t / CC, c = t % CC;
  const float* g = gates + (size_t)b * (4 * CC);
  float ig = sigm(g[c]);
  float fg = sigm(g[CC + c]);
  float gg = tanhf(g[2 * CC + c]);
  float og = sigm(g[3 * CC + c]);
  float cn = fg * cst[t] + ig * gg;
  cst[t] = cn;
  hst[t] = og * tanhf(cn);
}
__global__ void s2s_energy(const float* __restrict__ x, const float* __restrict__ hst,
                           float* __restrict__ en) {
  int n = blockIdx.x * blockDim.x + threadIdx.x;
  if (n >= NN) return;
  int g = n / NPGC;
  const float* xn = x + (size_t)n * CC;
  const float* qg = hst + (size_t)g * CC;
  float s = 0.0f;
  for (int c = 0; c < CC; ++c) s += xn[c] * qg[c];
  en[n] = s;
}
__global__ void s2s_norm(float* __restrict__ en) {  // per-graph softmax in place
  int b = blockIdx.x * blockDim.x + threadIdx.x;
  if (b >= BB) return;
  float m = -3.4e38f;
  for (int i = 0; i < NPGC; ++i) m = fmaxf(m, en[b * NPGC + i]);
  if (!(m == m) || fabsf(m) > 3.0e38f) m = 0.0f;
  float s = 0.0f;
  for (int i = 0; i < NPGC; ++i) s += __expf(en[b * NPGC + i] - m);
  float inv = 1.0f / (s + 1e-16f);
  for (int i = 0; i < NPGC; ++i) en[b * NPGC + i] = __expf(en[b * NPGC + i] - m) * inv;
}
__global__ void s2s_readout(const float* __restrict__ x, const float* __restrict__ en,
                            const float* __restrict__ hst, float* __restrict__ qstar) {
  int t = blockIdx.x * blockDim.x + threadIdx.x;
  if (t >= BB * CC) return;
  int b = t / CC, c = t % CC;
  float r = 0.0f;
  for (int i = 0; i < NPGC; ++i) {
    int n = b * NPGC + i;
    r += en[n] * x[(size_t)n * CC + c];
  }
  qstar[(size_t)b * (2 * CC) + c]      = hst[t];
  qstar[(size_t)b * (2 * CC) + CC + c] = r;
}

// ---------------- reparameterize: z = eps*exp(logvar) + mu ----------------
__global__ void reparam(const float* __restrict__ eps, const float* __restrict__ mu,
                        const float* __restrict__ lv, float* __restrict__ z) {
  int i = blockIdx.x * blockDim.x + threadIdx.x;
  if (i >= BB * LATD) return;
  z[i] = eps[i] * __expf(lv[i]) + mu[i];
}

// ---------------- orchestration ----------------
extern "C" void kernel_launch(void* const* d_in, const int* in_sizes, int n_in,
                              void* d_out, int out_size, void* d_ws, size_t ws_size,
                              hipStream_t stream) {
  (void)in_sizes; (void)n_in; (void)out_size; (void)ws_size;

  const float* x    = (const float*)d_in[0];   // [N,30]
  const float* eatt = (const float*)d_in[1];   // [E,11]
  const int*   ei   = (const int*)  d_in[2];   // [2,E]
  const float* eps  = (const float*)d_in[4];   // [B,128]

  // workspace carve (256B aligned)
  char* base = (char*)d_ws;
  size_t off = 0;
  auto carveB = [&](size_t bytes) { char* p = base + off; off += (bytes + 255) & ~(size_t)255; return p; };
  auto carveF = [&](size_t n) { return (float*)carveB(n * 4); };
  auto carveH = [&](size_t n) { return (__bf16*)carveB(n * 2); };

  float* hA     = carveF((size_t)NN * CC);
  float* hB     = carveF((size_t)NN * CC);
  float* qb     = carveF((size_t)NN * HC);
  float* kb     = carveF((size_t)NN * HC);
  float* vb     = carveF((size_t)NN * HC);
  float* outacc = carveF((size_t)NN * HC);
  float* xr     = carveF((size_t)NN * CC);
  float* alpha  = carveF((size_t)EE * HH);
  unsigned* amaxkey = (unsigned*)carveF((size_t)NN * HH);
  float* denom  = carveF((size_t)NN * HH);
  float* stats  = carveF(2 * CC);
  float* qstar  = carveF((size_t)BB * 2 * CC);
  float* hst    = carveF((size_t)BB * CC);
  float* cst    = carveF((size_t)BB * CC);
  float* gates  = carveF((size_t)BB * 4 * CC);
  float* en     = carveF((size_t)NN);
  float* zb     = carveF((size_t)BB * LATD);
  float* d1     = carveF((size_t)BB * DHD);
  float* d2     = carveF((size_t)BB * DHD);
  __bf16* abf   = carveH((size_t)NN * CC);     // activation bf16 scratch (max M*Kp)
  __bf16* wscr  = carveH((size_t)EOUT * DHD);  // weight bf16 scratch (max N*Kp)

  auto zero = [&](void* p, long words) {
    long blocks = (words + 255) / 256; if (blocks > 2048) blocks = 2048;
    zero_u32<<<dim3((unsigned)blocks), dim3(256), 0, stream>>>((unsigned*)p, words);
  };
  auto cvtR = [&](const float* s, __bf16* d, long M, int K, int Kp) {
    long tot = M * Kp;
    cvt_rows<<<dim3((unsigned)((tot + 255) / 256)), dim3(256), 0, stream>>>(s, d, M, K, Kp);
  };
  auto cvtT = [&](const float* s, __bf16* d, int K, int N, int Kp) {
    long tot = (long)N * Kp;
    cvt_wT<<<dim3((unsigned)((tot + 255) / 256)), dim3(256), 0, stream>>>(s, d, K, N, Kp);
  };
  auto gemm = [&](const __bf16* A, const __bf16* W, const float* bias, float* C,
                  int M, int N, int Kp, int flags) {
    int tiles = (M / 16) * ((N + 63) / 64);
    gemm_bf16<<<dim3((tiles + 3) / 4), dim3(128), 0, stream>>>(A, W, bias, C, M, N, Kp, flags);
  };
  // convert weight (row-major [K,N]) into wscr then GEMM with it
  auto gemmW = [&](const __bf16* A, const float* Wf, const float* bias, float* C,
                   int M, int N, int K, int Kp, int flags) {
    cvtT(Wf, wscr, K, N, Kp);
    gemm(A, wscr, bias, C, M, N, Kp, flags);
  };

  const int eBlocks = (EE * HH + 255) / 256;

  auto conv = [&](const float* hin, int Kin, int Kp, int pb, float* hout) {
    const float* Wq = (const float*)d_in[pb + 0]; const float* bq = (const float*)d_in[pb + 1];
    const float* Wk = (const float*)d_in[pb + 2]; const float* bk = (const float*)d_in[pb + 3];
    const float* Wv = (const float*)d_in[pb + 4]; const float* bv = (const float*)d_in[pb + 5];
    const float* We = (const float*)d_in[pb + 6];
    const float* Ws = (const float*)d_in[pb + 7]; const float* bs = (const float*)d_in[pb + 8];
    const float* Wb = (const float*)d_in[pb + 9];
    cvtR(hin, abf, NN, Kin, Kp);
    gemmW(abf, Wq, bq, qb, NN, HC, Kin, Kp, 0);
    gemmW(abf, Wk, bk, kb, NN, HC, Kin, Kp, 0);
    gemmW(abf, Wv, bv, vb, NN, HC, Kin, Kp, 0);
    gemmW(abf, Ws, bs, xr, NN, CC, Kin, Kp, 0);
    zero(amaxkey, (long)NN * HH);
    zero(denom,   (long)NN * HH);
    zero(outacc,  (long)NN * HC);
    edge_alpha    <<<dim3(eBlocks), dim3(256), 0, stream>>>(qb, kb, eatt, We, ei, alpha, amaxkey);
    edge_softmax  <<<dim3(eBlocks), dim3(256), 0, stream>>>(alpha, amaxkey, denom, ei);
    edge_aggregate<<<dim3(eBlocks), dim3(256), 0, stream>>>(alpha, denom, vb, eatt, We, ei, outacc);
    beta_combine  <<<dim3((NN + 127) / 128), dim3(128), 0, stream>>>(outacc, xr, Wb, hout);
  };
  auto bnorm = [&](float* h, int gIdx) {
    bn_stats<<<dim3(CC), dim3(256), 0, stream>>>(h, stats);
    bn_apply<<<dim3((unsigned)(((long)NN * CC + 255) / 256)), dim3(256), 0, stream>>>(
        h, stats, (const float*)d_in[gIdx], (const float*)d_in[gIdx + 1]);
  };

  // encoder: conv->relu->bn x3, conv->relu
  conv(x,  30, 32, 5,  hA); bnorm(hA, 45);
  conv(hA, CC, CC, 15, hB); bnorm(hB, 47);
  conv(hB, CC, CC, 25, hA); bnorm(hA, 49);
  conv(hA, CC, CC, 35, hB);
  const float* hfin = hB;

  // Set2Set (weights are [4C, in] row-major == col-major for our GEMM: use cvt_rows)
  const float* W_ih = (const float*)d_in[51];  // [256,128]
  const float* W_hh = (const float*)d_in[52];  // [256,64]
  const float* b_ih = (const float*)d_in[53];
  const float* b_hh = (const float*)d_in[54];
  zero(qstar, (long)BB * 2 * CC);
  zero(hst,   (long)BB * CC);
  zero(cst,   (long)BB * CC);
  for (int step = 0; step < 4; ++step) {
    cvtR(qstar, abf, BB, 2 * CC, 2 * CC);
    cvtR(W_ih, wscr, 4 * CC, 2 * CC, 2 * CC);
    gemm(abf, wscr, b_ih, gates, BB, 4 * CC, 2 * CC, 0);       // q_star @ W_ih^T
    cvtR(hst, abf, BB, CC, CC);
    cvtR(W_hh, wscr, 4 * CC, CC, CC);
    gemm(abf, wscr, b_hh, gates, BB, 4 * CC, CC, 2);           // += h @ W_hh^T
    lstm_cell  <<<dim3((BB * CC + 255) / 256), dim3(256), 0, stream>>>(gates, hst, cst);
    s2s_energy <<<dim3((NN + 255) / 256), dim3(256), 0, stream>>>(hfin, hst, en);
    s2s_norm   <<<dim3((BB + 255) / 256), dim3(256), 0, stream>>>(en);
    s2s_readout<<<dim3((BB * CC + 255) / 256), dim3(256), 0, stream>>>(hfin, en, hst, qstar);
  }

  // decoder + outputs: [triu(B*6125), node(B*500), mu(B*128), logvar(B*128)]
  float* out_triu = (float*)d_out;
  float* out_node = out_triu + (size_t)BB * EOUT;
  float* out_mu   = out_node + (size_t)BB * AOUT;
  float* out_lv   = out_mu   + (size_t)BB * LATD;

  cvtR(qstar, abf, BB, 2 * CC, 2 * CC);
  gemmW(abf, (const float*)d_in[55], (const float*)d_in[56], out_mu, BB, LATD, 2 * CC, 2 * CC, 0);
  gemmW(abf, (const float*)d_in[57], (const float*)d_in[58], out_lv, BB, LATD, 2 * CC, 2 * CC, 0);
  reparam<<<dim3((BB * LATD + 255) / 256), dim3(256), 0, stream>>>(eps, out_mu, out_lv, zb);

  cvtR(zb, abf, BB, LATD, LATD);
  gemmW(abf, (const float*)d_in[59], (const float*)d_in[60], d1, BB, DHD, LATD, LATD, 1);   // relu
  cvtR(d1, abf, BB, DHD, DHD);
  gemmW(abf, (const float*)d_in[61], (const float*)d_in[62], d2, BB, DHD, DHD, DHD, 1);     // relu
  cvtR(d2, abf, BB, DHD, DHD);
  gemmW(abf, (const float*)d_in[63], (const float*)d_in[64], out_node, BB, AOUT, DHD, DHD, 0);
  gemmW(abf, (const float*)d_in[65], (const float*)d_in[66], out_triu, BB, EOUT, DHD, DHD, 0);
}
